// SelfAttention_75316546503157
// MI455X (gfx1250) — matrix operand order repaired and checked by reading, not compiled, not run on hardware
//
#include <hip/hip_runtime.h>

typedef _Float16 f16;
typedef __attribute__((ext_vector_type(16))) _Float16 v16h;
typedef __attribute__((ext_vector_type(8)))  _Float16 v8h;
typedef __attribute__((ext_vector_type(8)))  float    v8f;
typedef unsigned int u32x4 __attribute__((ext_vector_type(4)));
typedef int          i32x4 __attribute__((ext_vector_type(4)));
typedef int          i32x8 __attribute__((ext_vector_type(8)));

#define Bb 2
#define Ss 2048
#define Hh 1024
#define NHh 16
#define DHh 64

__device__ __forceinline__ v16h cat16(v8h lo, v8h hi) {
  v16h r;
#pragma unroll
  for (int i = 0; i < 8; ++i) { r[i] = lo[i]; r[i + 8] = hi[i]; }
  return r;
}
__device__ __forceinline__ v8h ld8h(const f16* p) { return *(const v8h*)p; }

// TDM: DMA one linear 8KB block (1024 x 8B) global -> LDS.  D# layout per
// CDNA5 ISA 8.3/8.4: group0 = {count|gather, lds_addr, global_addr, type=2},
// group1 = {data_size=8B, tensor_dim0=1024, tensor_dim1=1, tile_dim0=1024}.
__device__ __forceinline__ void tdm_load_8kb(const f16* gsrc, f16* lds_dst) {
  const unsigned long long ga = (unsigned long long)(uintptr_t)gsrc;
  const unsigned int       la = (unsigned int)(uintptr_t)lds_dst;  // flat low 32 = LDS offset
  u32x4 g0;
  g0[0] = 1u;                                                  // count=1 (valid user D#)
  g0[1] = la;                                                  // lds_addr
  g0[2] = (unsigned int)(ga & 0xffffffffu);                    // global_addr[31:0]
  g0[3] = (unsigned int)((ga >> 32) & 0x01ffffffu) | (2u << 30);  // addr[56:32] | type=2
  i32x8 g1;
  g1[0] = (3 << 16);                 // workgroup_mask=0, data_size=3 (8 bytes)
  g1[1] = (int)(1024u << 16);        // tensor_dim0[15:0]=1024 (bits 63:48)
  g1[2] = (int)(1u << 16);           // tensor_dim0[31:16]=0, tensor_dim1[15:0]=1
  g1[3] = (int)(1024u << 16);        // tensor_dim1[31:16]=0, tile_dim0=1024
  g1[4] = 0;                         // tile_dim1=0 (1-D tile), tile_dim2=0
  g1[5] = 1024;                      // tensor_dim0_stride[31:0]
  g1[6] = 0;
  g1[7] = 0;
  i32x4 z4 = {};
#if __clang_major__ >= 23
  i32x8 z8 = {};
  __builtin_amdgcn_tensor_load_to_lds(g0, g1, z4, z4, z8, 0);
#else
  __builtin_amdgcn_tensor_load_to_lds(g0, g1, z4, z4, 0);
#endif
}

// ---------------------------------------------------------------------------
// Kernel 1: fused QKV projection.  X[4096x1024] fp32 -> Q/K/V f16 [B*NH][S][DH]
// Block = 256 threads (8 waves), tile = 128 rows x 64 cols, K-chunks of 32.
// Each wave owns a 64-row x 16-col strip = 4 subtiles x 3 matrices = 12 WMMA/chunk.
// ---------------------------------------------------------------------------
__global__ __launch_bounds__(256) void qkv_proj_kernel(
    const float* __restrict__ X,
    const float* __restrict__ Wq, const float* __restrict__ bq,
    const float* __restrict__ Wk, const float* __restrict__ bk,
    const float* __restrict__ Wv, const float* __restrict__ bv,
    f16* __restrict__ Qo, f16* __restrict__ Ko, f16* __restrict__ Vo)
{
  __shared__ __align__(16) f16 Xs[128 * 32];       // [row][k]   (A layout friendly)
  __shared__ __align__(16) f16 Ws[3][64 * 32];     // transposed [n][k] (B layout friendly)

  const int t     = threadIdx.x;
  const int wave  = t >> 5;
  const int lane  = t & 31;
  const int col16 = lane & 15;
  const int hi    = lane >> 4;

  const int rowbase = blockIdx.y * 128;  // output row tile (0..4095)
  const int colbase = blockIdx.x * 64;   // output col tile (0..1023)
  const int rg = (wave >> 2) * 64;       // wave row group: 0 or 64 (4 subtiles of 16)
  const int cb = (wave & 3) * 16;        // wave col sub-tile: 0,16,32,48

  const float* Wm[3] = {Wq, Wk, Wv};

  v8f acc[3][4] = {};

  for (int kc = 0; kc < Hh; kc += 32) {
    __syncthreads();
    { // X tile 128x32 f32 -> f16, natural layout (16 contiguous f32 per thread)
      const int rl = t >> 1;
      const int klx = (t & 1) * 16;
      const float* src = X + (size_t)(rowbase + rl) * Hh + kc + klx;
      v8h t0, t1;
#pragma unroll
      for (int j = 0; j < 8; ++j) { t0[j] = (f16)src[j]; t1[j] = (f16)src[8 + j]; }
      *(v8h*)&Xs[rl * 32 + klx] = t0;
      *(v8h*)&Xs[rl * 32 + klx + 8] = t1;
    }
    { // W tiles 32x64 f32, transposed into LDS as [n][k]
      const int kl = t >> 3;
      const int nl = (t & 7) * 8;
#pragma unroll
      for (int m = 0; m < 3; ++m) {
        const float* src = Wm[m] + (size_t)(kc + kl) * Hh + colbase + nl;
#pragma unroll
        for (int j = 0; j < 8; ++j) Ws[m][(nl + j) * 32 + kl] = (f16)src[j];
      }
    }
    __syncthreads();

    // A fragments: rows rg+16*i+col16; ISA A-layout k = e + 8*((e>>3)+hi)
    v16h a[4];
#pragma unroll
    for (int i = 0; i < 4; ++i) {
      const int row = rg + 16 * i + col16;
      a[i] = cat16(ld8h(&Xs[row * 32 + 8 * hi]),
                   ld8h(&Xs[row * 32 + 8 * hi + 16]));
    }
#pragma unroll
    for (int m = 0; m < 3; ++m) {
      // B fragment: col n = cb+col16, k = e + 8*((e>>3)+hi) (matches A)
      v16h bf = cat16(ld8h(&Ws[m][(cb + col16) * 32 + 8 * hi]), ld8h(&Ws[m][(cb + col16) * 32 + 8 * hi + 16]));
#pragma unroll
      for (int i = 0; i < 4; ++i)
        acc[m][i] = __builtin_amdgcn_wmma_f32_16x16x32_f16(false, a[i], false, bf,
                                                           (short)0, acc[m][i], false, false);
    }
  }

  // Epilogue: add bias, scale Q by 1/sqrt(DH), store f16 into [B*NH][S][DH]
  const int n_global = colbase + cb + col16;
  const int h  = n_global >> 6;
  const int dh = n_global & 63;
  const float biasv[3] = {bq[n_global], bk[n_global], bv[n_global]};
  f16* outp[3] = {Qo, Ko, Vo};
#pragma unroll
  for (int m = 0; m < 3; ++m) {
#pragma unroll
    for (int i = 0; i < 4; ++i) {
#pragma unroll
      for (int r = 0; r < 8; ++r) {
        const int mrow = rowbase + rg + 16 * i + r + 8 * hi;  // D row = r + 8*hi
        const int bidx = mrow >> 11;
        const int s    = mrow & 2047;
        float val = acc[m][i][r] + biasv[m];
        if (m == 0) val *= 0.125f;  // 1/sqrt(64)
        outp[m][(((size_t)(bidx * NHh + h)) * Ss + s) * DHh + dh] = (f16)val;
      }
    }
  }
}

// ---------------------------------------------------------------------------
// Kernel 2: flash attention.  Per block: one (b,h), 128 q rows (16 per wave).
// 64 keys/iter, double-buffered LDS.  K tiles arrive via TDM (tensor_load_to_lds,
// issued by wave 0, synced with s_wait_tensorcnt before the barrier); V tiles via
// register prefetch + transposed stash.  16 WMMAs per wave per iteration.
// ---------------------------------------------------------------------------
__global__ __launch_bounds__(256) void attn_kernel(
    const f16* __restrict__ Q, const f16* __restrict__ K, const f16* __restrict__ V,
    const float* __restrict__ mask, float* __restrict__ out)
{
  __shared__ __align__(16) f16 Ksh[2][64 * 64];   // [key][dh]   (TDM destination)
  __shared__ __align__(16) f16 Vsh[2][64 * 64];   // transposed [dh][key]
  __shared__ __align__(16) f16 Psh[8][16 * 64];   // per-wave P staging [q][key]

  const int t     = threadIdx.x;
  const int wave  = t >> 5;
  const int lane  = t & 31;
  const int col16 = lane & 15;
  const int hi    = lane >> 4;

  const int bh = blockIdx.y;            // 0..31
  const int b  = bh >> 4;
  const int h  = bh & 15;
  const int qbase = blockIdx.x * 128 + wave * 16;
  const size_t base = (size_t)bh * Ss * DHh;

  // V tile-loader mapping: 4 threads per key row, 16 halves each
  const int kl = t >> 2;           // key 0..63
  const int d  = (t & 3) * 16;     // dh 0,16,32,48

  // Q fragments: two dh-32 chunks, A layout
  v16h qa[2];
#pragma unroll
  for (int c = 0; c < 2; ++c) {
    const f16* qp = Q + base + (size_t)(qbase + col16) * DHh + 32 * c + 8 * hi;
    qa[c] = cat16(ld8h(qp), ld8h(qp + 16));
  }

  v8f o[4] = {};
  float mrow[8], lrow[8];
#pragma unroll
  for (int r = 0; r < 8; ++r) { mrow[r] = -1e30f; lrow[r] = 0.f; }

  const float* mk = mask + b * Ss;

  v8h vr0, vr1;  // V prefetch registers
  auto prefetchV = [&](int kb) {
    const size_t off = base + (size_t)(kb * 64 + kl) * DHh + d;
    vr0 = ld8h(V + off);  vr1 = ld8h(V + off + 8);
  };
  auto stashV = [&](int buf) {
#pragma unroll
    for (int j = 0; j < 8; ++j) {
      Vsh[buf][(d + j) * 64 + kl]     = vr0[j];
      Vsh[buf][(d + 8 + j) * 64 + kl] = vr1[j];
    }
  };

  const int NKB = Ss / 64;
  if (wave == 0) tdm_load_8kb(K + base, &Ksh[0][0]);   // K tile 0 via TDM
  prefetchV(0);
  stashV(0);

  for (int kb = 0; kb < NKB; ++kb) {
    const int cur = kb & 1;
    const bool more = (kb + 1) < NKB;
    if (more) prefetchV(kb + 1);            // registers only; hidden under compute
    __builtin_amdgcn_s_wait_tensorcnt(0);   // wave0: K tile kb landed (NOP elsewhere)
    __syncthreads();                        // buf[cur] ready; buf[cur^1] free
    if (more && wave == 0)                  // DMA next K tile into the free buffer
      tdm_load_8kb(K + base + (size_t)(kb + 1) * 64 * DHh, &Ksh[cur ^ 1][0]);

    // scores: 4 key tiles x 2 dh chunks = 8 WMMAs
    v8f s[4] = {};
#pragma unroll
    for (int c = 0; c < 2; ++c) {
#pragma unroll
      for (int n = 0; n < 4; ++n) {
        v16h kf = cat16(ld8h(&Ksh[cur][(16 * n + col16) * 64 + 32 * c + 8 * hi]), ld8h(&Ksh[cur][(16 * n + col16) * 64 + 32 * c + 8 * hi + 16]));
        s[n] = __builtin_amdgcn_wmma_f32_16x16x32_f16(false, qa[c], false, kf,
                                                      (short)0, s[n], false, false);
      }
    }

    float mv[4];
#pragma unroll
    for (int j = 0; j < 4; ++j) mv[j] = mk[kb * 64 + 16 * j + col16];

    float af[8];
#pragma unroll
    for (int r = 0; r < 8; ++r) {
      float x[4];
#pragma unroll
      for (int j = 0; j < 4; ++j) x[j] = s[j][r] + mv[j];
      float vmax = fmaxf(fmaxf(x[0], x[1]), fmaxf(x[2], x[3]));
#pragma unroll
      for (int m = 1; m < 16; m <<= 1) vmax = fmaxf(vmax, __shfl_xor(vmax, m, 32));
      const float mnew  = fmaxf(mrow[r], vmax);
      const float alpha = __expf(mrow[r] - mnew);
      float p[4], ps = 0.f;
#pragma unroll
      for (int j = 0; j < 4; ++j) { p[j] = __expf(x[j] - mnew); ps += p[j]; }
#pragma unroll
      for (int m = 1; m < 16; m <<= 1) ps += __shfl_xor(ps, m, 32);
      lrow[r] = lrow[r] * alpha + ps;
      mrow[r] = mnew;
      af[r]   = alpha;
      const int qr = r + 8 * hi;                  // D row -> P row
#pragma unroll
      for (int j = 0; j < 4; ++j) Psh[wave][qr * 64 + 16 * j + col16] = (f16)p[j];
    }
#pragma unroll
    for (int c = 0; c < 4; ++c)
#pragma unroll
      for (int r = 0; r < 8; ++r) o[c][r] *= af[r];

    // re-read P as A fragments (per-wave region; LDS in-order within wave)
    v16h pf[2];
#pragma unroll
    for (int pc = 0; pc < 2; ++pc) {
      const f16* pp = &Psh[wave][col16 * 64 + 32 * pc + 8 * hi];
      pf[pc] = cat16(ld8h(pp), ld8h(pp + 16));
    }
    // P V: 2 key chunks x 4 dh tiles = 8 WMMAs
#pragma unroll
    for (int pc = 0; pc < 2; ++pc) {
#pragma unroll
      for (int c = 0; c < 4; ++c) {
        v16h vf = cat16(ld8h(&Vsh[cur][(16 * c + col16) * 64 + 32 * pc + 8 * hi]), ld8h(&Vsh[cur][(16 * c + col16) * 64 + 32 * pc + 8 * hi + 16]));
        o[c] = __builtin_amdgcn_wmma_f32_16x16x32_f16(false, pf[pc], false, vf,
                                                      (short)0, o[c], false, false);
      }
    }

    if (more) stashV(cur ^ 1);            // safe: readers of cur^1 all passed barrier
  }

  // normalize and write fp32 [B,S,H]
#pragma unroll
  for (int r = 0; r < 8; ++r) {
    const int q = qbase + r + 8 * hi;
    const float inv = 1.0f / fmaxf(lrow[r], 1e-20f);
#pragma unroll
    for (int c = 0; c < 4; ++c)
      out[((size_t)(b * Ss + q)) * Hh + h * DHh + 16 * c + col16] = o[c][r] * inv;
  }
}

extern "C" void kernel_launch(void* const* d_in, const int* in_sizes, int n_in,
                              void* d_out, int out_size, void* d_ws, size_t ws_size,
                              hipStream_t stream) {
  const float* X    = (const float*)d_in[0];
  const float* mask = (const float*)d_in[1];
  const float* Wq   = (const float*)d_in[2];
  const float* bq   = (const float*)d_in[3];
  const float* Wk   = (const float*)d_in[4];
  const float* bk   = (const float*)d_in[5];
  const float* Wv   = (const float*)d_in[6];
  const float* bv   = (const float*)d_in[7];
  float* out = (float*)d_out;

  f16* Qw = (f16*)d_ws;                               // 8 MB each, 24 MB total
  f16* Kw = Qw + (size_t)Bb * NHh * Ss * DHh;
  f16* Vw = Kw + (size_t)Bb * NHh * Ss * DHh;

  dim3 gridA(Hh / 64, (Bb * Ss) / 128);               // 16 x 32 blocks
  qkv_proj_kernel<<<gridA, 256, 0, stream>>>(X, Wq, bq, Wk, bk, Wv, bv, Qw, Kw, Vw);

  dim3 gridB(Ss / 128, Bb * NHh);                     // 16 x 32 blocks
  attn_kernel<<<gridB, 256, 0, stream>>>(Qw, Kw, Vw, mask, out);
}
